// LSTM_Transformer_SparseAttention_12317966205232
// MI455X (gfx1250) — matrix operand order repaired and checked by reading, not compiled, not run on hardware
//
#include <hip/hip_runtime.h>
#include <hip/hip_bf16.h>
#include <math.h>

// ---------------------------------------------------------------------------
// LSTM(2) -> sparse attention -> 2x transformer encoder -> FC, on gfx1250.
// All GEMM-shaped math runs on v_wmma_f32_16x16x32_bf16 (bf16 in, f32 acc).
// ---------------------------------------------------------------------------

typedef __attribute__((ext_vector_type(16))) __bf16 v16bf;
typedef __attribute__((ext_vector_type(8)))  float  v8f;

#define HDIM   96
#define NHEAD  4
#define HEADD  24
#define TSEQ   1024
#define BATCH  16
#define BT     (BATCH * TSEQ)
#define G4H    384
#define FFD    192
#define LN_EPS 1e-5f

// --- WMMA fragment lane swizzles (CDNA5 ISA 7.12.2, 16-bit layouts) --------
// A 16x32 (MxK): lanes 0-15 = rows, lanes 16-31 = same rows, other K half.
__device__ __forceinline__ int a_swz(int e, int half) {
  return (e < 8) ? (half * 8 + e) : (16 + half * 8 + (e - 8));
}
// B 32x16 (KxN): lanes 0-15 = cols, lanes 16-31 = same cols, other K half (same K order as A).
__device__ __forceinline__ int b_swz(int e, int half) {
  return (e < 8) ? (half * 8 + e) : (16 + half * 8 + (e - 8));
}

__device__ __forceinline__ float sigmoidf_(float x) {
  return 1.f / (1.f + __expf(-x));
}
__device__ __forceinline__ unsigned int sortkey(float f) {
  unsigned int u = __float_as_uint(f);
  return (u & 0x80000000u) ? ~u : (u | 0x80000000u);
}

// ---------------------------------------------------------------------------
// Generic C[M,N] = act(A[M,K] @ W[N,K]^T + bias). A,W bf16 row-major.
// One wave per 16x16 output tile. M,N multiples of 16; K multiple of 32.
// ---------------------------------------------------------------------------
__global__ void __launch_bounds__(32)
gemm_bf16_wmma(const __bf16* __restrict__ A, const __bf16* __restrict__ W,
               const float* __restrict__ bias, float* __restrict__ C,
               __bf16* __restrict__ Cbf, int M, int N, int K, int relu) {
  const int lane = threadIdx.x & 31;
  const int l = lane & 15, half = lane >> 4;
  const int tm = blockIdx.x, tn = blockIdx.y;
  const size_t arow = (size_t)(tm * 16 + l) * K;
  const size_t wrow = (size_t)(tn * 16 + l) * K;
  v8f acc = {};
  for (int k0 = 0; k0 < K; k0 += 32) {
    const __bf16* Ap = A + arow + k0;
    const __bf16* Wp = W + wrow + k0;
    __builtin_prefetch(Ap + 32, 0, 0);  // global_prefetch_b8 for next K tile
    v16bf a, b;
#pragma unroll
    for (int e = 0; e < 16; ++e) a[e] = Ap[a_swz(e, half)];
#pragma unroll
    for (int e = 0; e < 16; ++e) b[e] = Wp[b_swz(e, half)];
    acc = __builtin_amdgcn_wmma_f32_16x16x32_bf16(false, a, false, b,
                                                  (short)0, acc, false, false);
  }
  const int n = tn * 16 + l;
  const float bn = bias ? bias[n] : 0.f;
#pragma unroll
  for (int j = 0; j < 8; ++j) {
    const int m = tm * 16 + half * 8 + j;
    float v = acc[j] + bn;
    if (relu) v = fmaxf(v, 0.f);
    C[(size_t)m * N + n] = v;
    if (Cbf) Cbf[(size_t)m * N + n] = (__bf16)v;
  }
}

// ---------------------------------------------------------------------------
// Persistent single-workgroup LSTM recurrence. gates_x = x @ w_ih^T is
// precomputed; per step we do h(16x96) @ w_hh^T(96x384) with WMMA out of LDS.
// LDS: w_hh bf16 (73728) + h bf16 (3072) + c (6144) + gates (24576) + bias.
// ---------------------------------------------------------------------------
__global__ void __launch_bounds__(128)
lstm_recurrence(const float* __restrict__ gates_x,  // [B*T, 384], row b*T+t
                const float* __restrict__ w_hh,     // [384, 96]
                const float* __restrict__ b_ih, const float* __restrict__ b_hh,
                __bf16* __restrict__ h_out) {       // [B*T, 96] bf16
  extern __shared__ char smem[];
  __bf16* wlds = (__bf16*)smem;                               // 384*96
  __bf16* hb   = (__bf16*)(smem + 73728);                     // 16*96
  float*  cst  = (float*)(smem + 73728 + 3072);               // 16*96
  float*  gts  = (float*)(smem + 73728 + 3072 + 6144);        // 16*384
  float*  bsum = (float*)(smem + 73728 + 3072 + 6144 + 24576);// 384
  const int tid = threadIdx.x;
  for (int i = tid; i < 384 * 96; i += 128) wlds[i] = (__bf16)w_hh[i];
  for (int i = tid; i < 384; i += 128) bsum[i] = b_ih[i] + b_hh[i];
  for (int i = tid; i < 16 * 96; i += 128) { cst[i] = 0.f; hb[i] = (__bf16)0.f; }
  __syncthreads();
  const int wave = tid >> 5, lane = tid & 31;
  const int l = lane & 15, half = lane >> 4;
  for (int t = 0; t < TSEQ; ++t) {
    // h @ w_hh^T : M=16 (batch), K=96, N=384; 6 N-tiles per wave.
    for (int nt = wave * 6; nt < wave * 6 + 6; ++nt) {
      const int col = nt * 16 + l;
      v8f acc = {};
      for (int k0 = 0; k0 < 96; k0 += 32) {
        v16bf a, b;
#pragma unroll
        for (int e = 0; e < 16; ++e) a[e] = hb[l * 96 + k0 + a_swz(e, half)];
#pragma unroll
        for (int e = 0; e < 16; ++e) b[e] = wlds[col * 96 + k0 + b_swz(e, half)];
        acc = __builtin_amdgcn_wmma_f32_16x16x32_bf16(false, a, false, b,
                                                      (short)0, acc, false, false);
      }
#pragma unroll
      for (int j = 0; j < 8; ++j) gts[(half * 8 + j) * 384 + col] = acc[j];
    }
    __syncthreads();
    // Fused gate activations + state update (16*96 cells).
    for (int idx = tid; idx < 16 * 96; idx += 128) {
      const int bb = idx / 96, j = idx - bb * 96;
      const float* gx = gates_x + ((size_t)bb * TSEQ + t) * G4H;
      const float ig = sigmoidf_(gts[bb * 384 + j]       + gx[j]       + bsum[j]);
      const float fg = sigmoidf_(gts[bb * 384 + 96 + j]  + gx[96 + j]  + bsum[96 + j]);
      const float gg = tanhf(    gts[bb * 384 + 192 + j] + gx[192 + j] + bsum[192 + j]);
      const float og = sigmoidf_(gts[bb * 384 + 288 + j] + gx[288 + j] + bsum[288 + j]);
      const float cn = fg * cst[idx] + ig * gg;
      cst[idx] = cn;
      const float hn = og * tanhf(cn);
      hb[idx] = (__bf16)hn;
      h_out[((size_t)bb * TSEQ + t) * HDIM + j] = (__bf16)hn;
    }
    __syncthreads();
  }
}

// ---------------------------------------------------------------------------
// Attention: one workgroup per (16-query tile, head, batch). Scores built by
// WMMA into LDS; optional top-k (keep scores >= kth largest, ties kept, as in
// the reference) via radix binary search; softmax; P@V by WMMA with LDS
// reduction across waves. num_keep==0 => dense softmax (encoder MHA).
// q/k/v are f32 rows of stride `rowstride` (96 for sparse, 288 for fused qkv).
// ---------------------------------------------------------------------------
__global__ void __launch_bounds__(128)
attention_wmma(const float* __restrict__ qp, const float* __restrict__ kp,
               const float* __restrict__ vp, int rowstride,
               float* __restrict__ outf, __bf16* __restrict__ outbf,
               int num_keep) {
  extern __shared__ char smem[];
  float*  S   = (float*)smem;                    // 16*1024 f32  = 65536 B
  __bf16* P   = (__bf16*)(smem + 65536);         // 16*1024 bf16 = 32768 B
  float*  red = (float*)(smem + 65536 + 32768);  // 4*16*16 f32  =  4096 B
  const int qt = blockIdx.x, head = blockIdx.y, b = blockIdx.z;
  const int tid = threadIdx.x, wave = tid >> 5, lane = tid & 31;
  const int l = lane & 15, half = lane >> 4;
  const size_t base = (size_t)b * TSEQ;
  const int hoff = head * HEADD;
  const float scale = 0.2041241452f;  // 1/sqrt(24)

  // Q fragment (K-dim = HD=24, zero-padded to 32). Reused for all key tiles.
  v16bf aq;
#pragma unroll
  for (int e = 0; e < 16; ++e) {
    const int d = a_swz(e, half);
    aq[e] = (d < HEADD)
        ? (__bf16)qp[(base + qt * 16 + l) * rowstride + hoff + d] : (__bf16)0.f;
  }
  // Scores: 64 key tiles split across 4 waves.
  for (int kt = wave; kt < TSEQ / 16; kt += 4) {
    v16bf bk;
#pragma unroll
    for (int e = 0; e < 16; ++e) {
      const int d = b_swz(e, half);
      bk[e] = (d < HEADD)
          ? (__bf16)kp[(base + kt * 16 + l) * rowstride + hoff + d] : (__bf16)0.f;
    }
    v8f sc = {};
    sc = __builtin_amdgcn_wmma_f32_16x16x32_bf16(false, aq, false, bk,
                                                 (short)0, sc, false, false);
#pragma unroll
    for (int j = 0; j < 8; ++j) S[(half * 8 + j) * TSEQ + kt * 16 + l] = sc[j] * scale;
  }
  __syncthreads();

  // Per-row top-k threshold + softmax. One wave per row, 4 rows per wave.
  for (int r = wave; r < 16; r += 4) {
    float* Sr = S + r * TSEQ;
    unsigned int kth = 0u;
    if (num_keep > 0) {
      unsigned int lo = 0u, hi = 0xFFFFFFFFu;  // max u with count(key>=u) >= keep
      while (lo < hi) {
        const unsigned int mid = lo + ((hi - lo + 1u) >> 1);
        int cnt = 0;
        for (int jj = lane; jj < TSEQ; jj += 32) cnt += (sortkey(Sr[jj]) >= mid);
#pragma unroll
        for (int o = 16; o > 0; o >>= 1) cnt += __shfl_down(cnt, o, 32);
        cnt = __shfl(cnt, 0, 32);
        if (cnt >= num_keep) lo = mid; else hi = mid - 1u;
      }
      kth = lo;
    }
    float mx = -3.0e38f;
    for (int jj = lane; jj < TSEQ; jj += 32)
      if (sortkey(Sr[jj]) >= kth) mx = fmaxf(mx, Sr[jj]);
#pragma unroll
    for (int o = 16; o > 0; o >>= 1) mx = fmaxf(mx, __shfl_down(mx, o, 32));
    mx = __shfl(mx, 0, 32);
    float sum = 0.f;
    for (int jj = lane; jj < TSEQ; jj += 32) {
      const float e = (sortkey(Sr[jj]) >= kth) ? __expf(Sr[jj] - mx) : 0.f;
      Sr[jj] = e;
      sum += e;
    }
#pragma unroll
    for (int o = 16; o > 0; o >>= 1) sum += __shfl_down(sum, o, 32);
    sum = __shfl(sum, 0, 32);
    const float inv = 1.f / sum;
    for (int jj = lane; jj < TSEQ; jj += 32) P[r * TSEQ + jj] = (__bf16)(Sr[jj] * inv);
  }
  __syncthreads();

  // O(16x24) = P(16x1024) @ V(1024x24). Waves: (N-tile, K-half) grid 2x2.
  const int ntile = wave & 1, khalf = wave >> 1;
  v8f acc = {};
  for (int kk = khalf * 16; kk < khalf * 16 + 16; ++kk) {  // 32 keys/iter
    v16bf ap, bv;
#pragma unroll
    for (int e = 0; e < 16; ++e) ap[e] = P[l * TSEQ + kk * 32 + a_swz(e, half)];
#pragma unroll
    for (int e = 0; e < 16; ++e) {
      const int d = ntile * 16 + l;
      const int key = kk * 32 + b_swz(e, half);
      bv[e] = (d < HEADD)
          ? (__bf16)vp[(base + key) * rowstride + hoff + d] : (__bf16)0.f;
    }
    acc = __builtin_amdgcn_wmma_f32_16x16x32_bf16(false, ap, false, bv,
                                                  (short)0, acc, false, false);
  }
#pragma unroll
  for (int j = 0; j < 8; ++j) red[wave * 256 + (half * 8 + j) * 16 + l] = acc[j];
  __syncthreads();
  for (int idx = tid; idx < 512; idx += 128) {
    const int nt = idx >> 8, m = (idx >> 4) & 15, dl = idx & 15;
    const int d = nt * 16 + dl;
    if (d < HEADD) {
      const float v = red[nt * 256 + m * 16 + dl] + red[(2 + nt) * 256 + m * 16 + dl];
      const size_t o = (base + qt * 16 + m) * HDIM + hoff + d;
      outf[o] = v;
      if (outbf) outbf[o] = (__bf16)v;
    }
  }
}

// ---------------------------------------------------------------------------
// out = LayerNorm(a + r) * w + b ; one block per row of 96.
// ---------------------------------------------------------------------------
__global__ void __launch_bounds__(128)
add_layernorm(const float* __restrict__ a, const float* __restrict__ r,
              const float* __restrict__ w, const float* __restrict__ bb,
              float* __restrict__ outf, __bf16* __restrict__ outbf) {
  __shared__ float sd[128];
  const int row = blockIdx.x, tid = threadIdx.x;
  float v = 0.f;
  if (tid < HDIM) v = a[(size_t)row * HDIM + tid] + r[(size_t)row * HDIM + tid];
  sd[tid] = v;
  __syncthreads();
  for (int o = 64; o > 0; o >>= 1) { if (tid < o) sd[tid] += sd[tid + o]; __syncthreads(); }
  const float mu = sd[0] / HDIM;
  __syncthreads();
  const float d = (tid < HDIM) ? (v - mu) : 0.f;
  sd[tid] = d * d;
  __syncthreads();
  for (int o = 64; o > 0; o >>= 1) { if (tid < o) sd[tid] += sd[tid + o]; __syncthreads(); }
  const float var = sd[0] / HDIM;
  if (tid < HDIM) {
    const float y = (v - mu) * rsqrtf(var + LN_EPS) * w[tid] + bb[tid];
    outf[(size_t)row * HDIM + tid] = y;
    if (outbf) outbf[(size_t)row * HDIM + tid] = (__bf16)y;
  }
}

__global__ void cvt_f32_bf16(const float* __restrict__ in,
                             __bf16* __restrict__ out, size_t n) {
  size_t i = (size_t)blockIdx.x * blockDim.x + threadIdx.x;
  const size_t stride = (size_t)gridDim.x * blockDim.x;
  for (; i < n; i += stride) out[i] = (__bf16)in[i];
}

__global__ void __launch_bounds__(128)
final_fc(const float* __restrict__ x, const float* __restrict__ fw,
         const float* __restrict__ fb, float* __restrict__ out) {
  __shared__ float sd[128];
  const int b = blockIdx.x, tid = threadIdx.x;
  sd[tid] = (tid < HDIM)
      ? x[((size_t)b * TSEQ + (TSEQ - 1)) * HDIM + tid] * fw[tid] : 0.f;
  __syncthreads();
  for (int o = 64; o > 0; o >>= 1) { if (tid < o) sd[tid] += sd[tid + o]; __syncthreads(); }
  if (tid == 0) out[b] = sd[0] + fb[0];
}

// ---------------------------------------------------------------------------
extern "C" void kernel_launch(void* const* d_in, const int* in_sizes, int n_in,
                              void* d_out, int out_size, void* d_ws, size_t ws_size,
                              hipStream_t stream) {
  (void)in_sizes; (void)n_in; (void)out_size; (void)ws_size;
  const float* x      = (const float*)d_in[0];
  const float* w_ih0  = (const float*)d_in[1];
  const float* w_hh0  = (const float*)d_in[2];
  const float* b_ih0  = (const float*)d_in[3];
  const float* b_hh0  = (const float*)d_in[4];
  const float* w_ih1  = (const float*)d_in[5];
  const float* w_hh1  = (const float*)d_in[6];
  const float* b_ih1  = (const float*)d_in[7];
  const float* b_hh1  = (const float*)d_in[8];
  const float* qw     = (const float*)d_in[9];
  const float* kw     = (const float*)d_in[10];
  const float* vw     = (const float*)d_in[11];
  const float* ow     = (const float*)d_in[12];
  const float* enc_in_w  = (const float*)d_in[13];
  const float* enc_in_b  = (const float*)d_in[14];
  const float* enc_out_w = (const float*)d_in[15];
  const float* enc_out_b = (const float*)d_in[16];
  const float* enc_l1_w  = (const float*)d_in[17];
  const float* enc_l1_b  = (const float*)d_in[18];
  const float* enc_l2_w  = (const float*)d_in[19];
  const float* enc_l2_b  = (const float*)d_in[20];
  const float* ln1_w  = (const float*)d_in[21];
  const float* ln1_b  = (const float*)d_in[22];
  const float* ln2_w  = (const float*)d_in[23];
  const float* ln2_b  = (const float*)d_in[24];
  const float* fc_w   = (const float*)d_in[25];
  const float* fc_b   = (const float*)d_in[26];
  float* out = (float*)d_out;

  // Deterministic bump allocator over d_ws (256B aligned slots).
  char* ws = (char*)d_ws;
  size_t off = 0;
  auto take = [&](size_t bytes) -> char* {
    char* p = ws + off;
    off += (bytes + 255) & ~(size_t)255;
    return p;
  };
  __bf16* x_bf     = (__bf16*)take((size_t)BT * 64 * 2);
  __bf16* wih0_bf  = (__bf16*)take((size_t)G4H * 64 * 2);
  __bf16* wih1_bf  = (__bf16*)take((size_t)G4H * 96 * 2);
  __bf16* qw_bf    = (__bf16*)take((size_t)96 * 96 * 2);
  __bf16* kw_bf    = (__bf16*)take((size_t)96 * 96 * 2);
  __bf16* vw_bf    = (__bf16*)take((size_t)96 * 96 * 2);
  __bf16* ow_bf    = (__bf16*)take((size_t)96 * 96 * 2);
  __bf16* encin_bf = (__bf16*)take((size_t)2 * 288 * 96 * 2);
  __bf16* encout_bf= (__bf16*)take((size_t)2 * 96 * 96 * 2);
  __bf16* l1_bf    = (__bf16*)take((size_t)2 * FFD * 96 * 2);
  __bf16* l2_bf    = (__bf16*)take((size_t)2 * 96 * FFD * 2);
  float*  gbuf     = (float*)take((size_t)BT * G4H * 4);   // gates_x / fused qkv
  __bf16* h0_bf    = (__bf16*)take((size_t)BT * 96 * 2);
  __bf16* h1_bf    = (__bf16*)take((size_t)BT * 96 * 2);
  float*  qf       = (float*)take((size_t)BT * 96 * 4);
  float*  kf       = (float*)take((size_t)BT * 96 * 4);
  float*  vf       = (float*)take((size_t)BT * 96 * 4);
  float*  ctx      = (float*)take((size_t)BT * 96 * 4);
  __bf16* ctx_bf   = (__bf16*)take((size_t)BT * 96 * 2);
  float*  xs       = (float*)take((size_t)BT * 96 * 4);    // encoder state
  __bf16* xs_bf    = (__bf16*)take((size_t)BT * 96 * 2);
  float*  x1       = (float*)take((size_t)BT * 96 * 4);
  __bf16* x1_bf    = (__bf16*)take((size_t)BT * 96 * 2);
  float*  ff1      = (float*)take((size_t)BT * FFD * 4);
  __bf16* ff1_bf   = (__bf16*)take((size_t)BT * FFD * 2);
  float*  ff2      = (float*)take((size_t)BT * 96 * 4);

  auto cvt = [&](const float* src, __bf16* dst, size_t n) {
    int blocks = (int)((n + 255) / 256); if (blocks > 2048) blocks = 2048;
    cvt_f32_bf16<<<blocks, 256, 0, stream>>>(src, dst, n);
  };
  auto gemm = [&](const __bf16* A, const __bf16* W, const float* bias, float* C,
                  __bf16* Cbf, int M, int N, int K, int relu) {
    gemm_bf16_wmma<<<dim3(M / 16, N / 16), 32, 0, stream>>>(A, W, bias, C, Cbf,
                                                            M, N, K, relu);
  };

  // --- one-shot bf16 conversions (weights are tiny; L2-resident) ----------
  cvt(x, x_bf, (size_t)BT * 64);
  cvt(w_ih0, wih0_bf, (size_t)G4H * 64);
  cvt(w_ih1, wih1_bf, (size_t)G4H * 96);
  cvt(qw, qw_bf, 96 * 96);
  cvt(kw, kw_bf, 96 * 96);
  cvt(vw, vw_bf, 96 * 96);
  cvt(ow, ow_bf, 96 * 96);
  cvt(enc_in_w, encin_bf, (size_t)2 * 288 * 96);
  cvt(enc_out_w, encout_bf, (size_t)2 * 96 * 96);
  cvt(enc_l1_w, l1_bf, (size_t)2 * FFD * 96);
  cvt(enc_l2_w, l2_bf, (size_t)2 * 96 * FFD);

  const size_t lstm_shmem = 73728 + 3072 + 6144 + 24576 + 1536;
  const size_t attn_shmem = 65536 + 32768 + 4096;

  // --- LSTM layer 0: batched input GEMM, then persistent recurrence -------
  gemm(x_bf, wih0_bf, nullptr, gbuf, nullptr, BT, G4H, 64, 0);
  lstm_recurrence<<<1, 128, lstm_shmem, stream>>>(gbuf, w_hh0, b_ih0, b_hh0, h0_bf);
  // --- LSTM layer 1 --------------------------------------------------------
  gemm(h0_bf, wih1_bf, nullptr, gbuf, nullptr, BT, G4H, 96, 0);
  lstm_recurrence<<<1, 128, lstm_shmem, stream>>>(gbuf, w_hh1, b_ih1, b_hh1, h1_bf);

  // --- sparse attention ----------------------------------------------------
  gemm(h1_bf, qw_bf, nullptr, qf, nullptr, BT, 96, 96, 0);
  gemm(h1_bf, kw_bf, nullptr, kf, nullptr, BT, 96, 96, 0);
  gemm(h1_bf, vw_bf, nullptr, vf, nullptr, BT, 96, 96, 0);
  attention_wmma<<<dim3(TSEQ / 16, NHEAD, BATCH), 128, attn_shmem, stream>>>(
      qf, kf, vf, HDIM, ctx, ctx_bf, TSEQ / 2);
  gemm(ctx_bf, ow_bf, nullptr, xs, xs_bf, BT, 96, 96, 0);

  // --- 2 transformer encoder layers ---------------------------------------
  for (int i = 0; i < 2; ++i) {
    gemm(xs_bf, encin_bf + (size_t)i * 288 * 96, enc_in_b + i * 288,
         gbuf, nullptr, BT, 288, 96, 0);  // fused qkv [BT, 288]
    attention_wmma<<<dim3(TSEQ / 16, NHEAD, BATCH), 128, attn_shmem, stream>>>(
        gbuf + 0, gbuf + 96, gbuf + 192, 288, ctx, ctx_bf, /*dense*/ 0);
    gemm(ctx_bf, encout_bf + (size_t)i * 96 * 96, enc_out_b + i * 96,
         ff2, nullptr, BT, 96, 96, 0);    // attn projection (temp in ff2)
    add_layernorm<<<BT, 128, 0, stream>>>(xs, ff2, ln1_w + i * 96, ln1_b + i * 96,
                                          x1, x1_bf);
    gemm(x1_bf, l1_bf + (size_t)i * FFD * 96, enc_l1_b + i * FFD,
         ff1, ff1_bf, BT, FFD, 96, 1);    // ReLU fused in epilogue
    gemm(ff1_bf, l2_bf + (size_t)i * 96 * FFD, enc_l2_b + i * 96,
         ff2, nullptr, BT, 96, FFD, 0);
    add_layernorm<<<BT, 128, 0, stream>>>(x1, ff2, ln2_w + i * 96, ln2_b + i * 96,
                                          xs, xs_bf);
  }

  // --- final FC over h[:, -1, :] ------------------------------------------
  final_fc<<<BATCH, 128, 0, stream>>>(xs, fc_w, fc_b, out);
}
